// Upsample_74259984548363
// MI455X (gfx1250) — matrix operand order repaired and checked by reading, not compiled, not run on hardware
//
#include <hip/hip_runtime.h>
#include <hip/hip_bf16.h>

#define C_CH   64
#define K_TAPS 27
#define LDS_ROW 72   // padded LDS row (bf16 units) = 144 B = 36 dwords -> 16 distinct banks

typedef __bf16 bf16_t;
typedef __attribute__((ext_vector_type(16))) __bf16 v16bf;
typedef __attribute__((ext_vector_type(8)))  float  v8f;

union ABFrag { v16bf v; uint4 q[2]; };

// ---- pass 1: x (f32) -> bf16 (x4 vectorized); appends one all-zero row at [n4..total4) ----
__global__ void cvt_x_bf16(const float* __restrict__ in, bf16_t* __restrict__ out,
                           int n4, int total4) {
    int i = blockIdx.x * blockDim.x + threadIdx.x;
    if (i >= total4) return;
    unsigned long long u = 0ull;
    if (i < n4) {
        float4 f = ((const float4*)in)[i];
        union { unsigned long long u; bf16_t h[4]; } p;
        p.h[0] = (bf16_t)f.x; p.h[1] = (bf16_t)f.y;
        p.h[2] = (bf16_t)f.z; p.h[3] = (bf16_t)f.w;
        u = p.u;
    }
    ((unsigned long long*)out)[i] = u;   // zero row for masked-out taps
}

// ---- pass 2: w[k][c][n] (f32) -> wT[k][n][c] (bf16) ----
__global__ void cvt_w_T(const float* __restrict__ w, bf16_t* __restrict__ wT) {
    int i = blockIdx.x * blockDim.x + threadIdx.x;   // i = k*4096 + n*64 + c
    if (i >= K_TAPS * C_CH * C_CH) return;
    int c = i & 63;
    int n = (i >> 6) & 63;
    int k = i >> 12;
    wT[i] = (bf16_t)w[(k << 12) + (c << 6) + n];
}

// ---- main: gather + 27-tap sparse conv as bf16 WMMA, f32 accumulate ----
// One wave processes FOUR 16-row tiles (64 output rows): per tap,
// 32 v_wmma : 16 ds_load_b128 : 16 A global_load_b128. Masking is folded into
// the gather index (zero row). The neigh->parent index chain for tap k+1 is
// issued before tap k's WMMA block so its latency hides under the matrix ops.
__global__ __launch_bounds__(256, 1) void octconv_wmma(
    const bf16_t* __restrict__ x_bf,          // [N_IN+1][64] bf16 (last row = zeros)
    const bf16_t* __restrict__ wT,            // [27][64][64] bf16, n-major
    const int* __restrict__ parent_idx,       // [N_OUT]
    const int* __restrict__ neigh,            // [N_OUT][27]
    const unsigned char* __restrict__ nmask,  // [N_OUT][27] (bool bytes)
    const float* __restrict__ bias,           // [64]
    float* __restrict__ out,                  // [N_OUT][64] f32
    int n_out, int zrow)                      // zrow = index of zero row
{
    extern __shared__ bf16_t lds_w[];         // [27*64][LDS_ROW]

    // Stage all 27 weight matrices into LDS with padded rows.
    {
        const uint4* src = (const uint4*)wT;              // 8x uint4 per 64-bf16 row
        for (int i = threadIdx.x; i < K_TAPS * C_CH * (C_CH / 8); i += blockDim.x) {
            int row = i >> 3, part = i & 7;
            ((uint4*)(lds_w + row * LDS_ROW))[part] = src[i];
        }
    }
    __syncthreads();

    const int lane  = threadIdx.x & 31;
    const int wv    = threadIdx.x >> 5;
    const int blk64 = blockIdx.x * 8 + wv;    // 64-row block per wave
    if (blk64 * 64 >= n_out) return;

    const int r  = lane & 15;                 // row-in-tile (A) / column (B,D)
    const int hl = lane >> 4;                 // lane half

    const int base = blk64 * 64;
    int rowA[4];
#pragma unroll
    for (int t = 0; t < 4; ++t) rowA[t] = base + t * 16 + r;

    float bs[4];
#pragma unroll
    for (int nc = 0; nc < 4; ++nc) bs[nc] = bias[nc * 16 + r];

    v8f acc[4][4] = {};                       // 4 tiles x 4 N-chunks (128 VGPRs)

    // Prologue: resolve gather pointers for tap 0.
    const bf16_t* xr[4];
#pragma unroll
    for (int t = 0; t < 4; ++t) {
        const int nIdx       = neigh[rowA[t] * K_TAPS + 0];
        const unsigned int m = nmask[rowA[t] * K_TAPS + 0];
        int pj               = parent_idx[nIdx];
        pj = m ? pj : zrow;
        xr[t] = x_bf + (size_t)pj * C_CH;
    }

    for (int k = 0; k < K_TAPS; ++k) {
        // A fragments for tap k (addresses already resolved last iteration).
        // 16-bit A layout: half 0 -> K {c0..c0+7, c0+16..c0+23}, half 1 -> +8.
        ABFrag a[4][2];
#pragma unroll
        for (int t = 0; t < 4; ++t) {
#pragma unroll
            for (int kc = 0; kc < 2; ++kc) {
                const int c0 = kc * 32 + hl * 8;
                a[t][kc].q[0] = *(const uint4*)(xr[t] + c0);        // 8 bf16
                a[t][kc].q[1] = *(const uint4*)(xr[t] + c0 + 16);   // 8 bf16
            }
        }

        // Issue next tap's index chain now; its latency hides under the WMMAs.
        const int kn = (k < K_TAPS - 1) ? k + 1 : k;   // clamped (redundant last iter)
        const bf16_t* xrn[4];
#pragma unroll
        for (int t = 0; t < 4; ++t) {
            const int nIdx       = neigh[rowA[t] * K_TAPS + kn];
            const unsigned int m = nmask[rowA[t] * K_TAPS + kn];
            int pj               = parent_idx[nIdx];
            pj = m ? pj : zrow;
            xrn[t] = x_bf + (size_t)pj * C_CH;
        }

        // B fragments from LDS (16-bit B layout: half 0 -> K {0..7, 16..23}, half 1 -> +8);
        // each fragment feeds four independent accumulator chains.
        const bf16_t* wk = lds_w + k * C_CH * LDS_ROW;
#pragma unroll
        for (int nc = 0; nc < 4; ++nc) {
            const bf16_t* wrow = wk + (nc * 16 + r) * LDS_ROW + hl * 8;
#pragma unroll
            for (int kc = 0; kc < 2; ++kc) {
                ABFrag bfr;
                bfr.q[0] = *(const uint4*)(wrow + kc * 32);      // 8 bf16 (16 B); next 8 at +16
                bfr.q[1] = *(const uint4*)(wrow + kc * 32 + 16);
#pragma unroll
                for (int t = 0; t < 4; ++t) {
                    acc[t][nc] = __builtin_amdgcn_wmma_f32_16x16x32_bf16(
                        false, a[t][kc].v, false, bfr.v, (short)0, acc[t][nc], false, false);
                }
            }
        }

#pragma unroll
        for (int t = 0; t < 4; ++t) xr[t] = xrn[t];
    }

    // D layout: VGPR v <-> row M = hl*8+v, lane&15 <-> column N. Add bias, store f32.
#pragma unroll
    for (int t = 0; t < 4; ++t) {
#pragma unroll
        for (int nc = 0; nc < 4; ++nc) {
#pragma unroll
            for (int v = 0; v < 8; ++v) {
                const int orow = base + t * 16 + hl * 8 + v;
                out[(size_t)orow * C_CH + nc * 16 + r] = acc[t][nc][v] + bs[nc];
            }
        }
    }
}

extern "C" void kernel_launch(void* const* d_in, const int* in_sizes, int n_in_args,
                              void* d_out, int out_size, void* d_ws, size_t ws_size,
                              hipStream_t stream) {
    const float*         x          = (const float*)d_in[0];
    const int*           parent_idx = (const int*)d_in[1];
    const int*           neigh      = (const int*)d_in[2];
    const unsigned char* nmask      = (const unsigned char*)d_in[3]; // jax bool -> 1 byte
    const float*         w          = (const float*)d_in[4];
    const float*         b          = (const float*)d_in[5];
    float*               out        = (float*)d_out;

    const int n_in  = in_sizes[0] / C_CH;   // 131072
    const int n_out = in_sizes[1];          // 1048576

    bf16_t* x_bf = (bf16_t*)d_ws;           // (n_in + 1) rows; last row zeros
    bf16_t* wT   = (bf16_t*)((char*)d_ws + (size_t)(n_in + 1) * C_CH * sizeof(bf16_t));

    {   // convert x to bf16 + append zero row
        int n4     = n_in * C_CH / 4;
        int total4 = (n_in + 1) * C_CH / 4;
        cvt_x_bf16<<<(total4 + 255) / 256, 256, 0, stream>>>(x, x_bf, n4, total4);
    }
    {   // convert + transpose weights
        int n = K_TAPS * C_CH * C_CH;
        cvt_w_T<<<(n + 255) / 256, 256, 0, stream>>>(w, wT);
    }

    const int blk64s = (n_out + 63) / 64;
    const int blocks = (blk64s + 7) / 8;    // 8 waves (64-row blocks) per 256-thread block
    const size_t lds_bytes = (size_t)K_TAPS * C_CH * LDS_ROW * sizeof(bf16_t); // 248832 B
    octconv_wmma<<<blocks, 256, lds_bytes, stream>>>(x_bf, wT, parent_idx, neigh,
                                                     nmask, b, out, n_out, n_in);
}
